// Head_54065048323007
// MI455X (gfx1250) — matrix operand order repaired and checked by reading, not compiled, not run on hardware
//
#include <hip/hip_runtime.h>
#include <hip/hip_bf16.h>

typedef __attribute__((ext_vector_type(16))) _Float16 v16h;
typedef __attribute__((ext_vector_type(8)))  _Float16 v8h;
typedef __attribute__((ext_vector_type(8)))  float    v8f;
typedef __attribute__((ext_vector_type(4)))  float    v4f;

#define N_EMBD 1024
#define HEAD   128
#define TSEQ   4096
#define NBATCH 4
#define TOKENS (NBATCH * TSEQ)   // 16384

// ---------------------------------------------------------------------------
// A-matrix 16x32 f16 fragment (contiguous k order used here; B matches):
//   lane m = lane&15 holds row M=m.
//   lo-half lanes: K 0..15 contiguous
//   hi-half lanes: K 16..31 contiguous
// base = start of this lane's row at the chunk's K offset (row-major f16).
__device__ __forceinline__ v16h load_fragA_f16(const _Float16* base, int hi) {
    return *(const v16h*)(base + hi * 16);
}

// B-matrix 32x16 f16 fragment: lane holds column N=lane&15; K contiguous,
// lo-half K 0..15, hi-half K 16..31 (same k order as A).
// rowbase = start of column's data in K-major-contiguous storage (i.e. the
// matrix stored transposed, row-major in K), at the chunk offset.
__device__ __forceinline__ v16h load_fragB_f16(const _Float16* rowbase, int hi) {
    return *(const v16h*)(rowbase + hi * 16);
}

// ---------------------------------------------------------------------------
__global__ void f32_to_f16_kernel(const float* __restrict__ src,
                                  _Float16* __restrict__ dst, int n) {
    int i = blockIdx.x * blockDim.x + threadIdx.x;
    if (i < n) dst[i] = (_Float16)src[i];
}

// ---------------------------------------------------------------------------
// One wave computes a 16-token x 128-col projection tile: out = x @ W^T + b.
// mode: 0 = Q (scaled by H^-1/2, row-major f16), 1 = K (row-major f16),
//       2 = V (stored transposed [batch][128][TSEQ] f16).
__global__ void qkv_proj_kernel(const float* __restrict__ x,
                                const _Float16* __restrict__ w16,   // [128,1024] f16
                                const float* __restrict__ bias,     // [128] f32
                                _Float16* __restrict__ outbuf,
                                int mode, float qscale) {
    const int lane = threadIdx.x & 31;
    const int wave = threadIdx.x >> 5;
    const int tile = blockIdx.x * 4 + wave;       // 0..1023 token tiles
    const int m    = lane & 15;
    const int hi   = lane >> 4;
    const int r0   = tile * 16;                   // first token of tile

    v8f acc[8] = {};

    const float* xrow = x + (size_t)(r0 + m) * N_EMBD;
    const _Float16* wrow = w16 + (size_t)m * N_EMBD;  // lane's column h-row base

#pragma unroll 1
    for (int cc = 0; cc < N_EMBD / 32; ++cc) {
        // ---- issue ALL loads for this chunk first (x + 8 W fragments), so the
        // compiler can clause them and drain loadcnt incrementally across WMMAs
        const float* xb = xrow + cc * 32 + hi * 16;
        v4f a0 = *(const v4f*)(xb);
        v4f a1 = *(const v4f*)(xb + 4);
        v4f a2 = *(const v4f*)(xb + 8);
        v4f a3 = *(const v4f*)(xb + 12);

        v16h bf[8];
#pragma unroll
        for (int ht = 0; ht < 8; ++ht)
            bf[ht] = load_fragB_f16(wrow + (size_t)ht * 16 * N_EMBD + cc * 32, hi);

        v16h a;
#pragma unroll
        for (int i = 0; i < 4; ++i) {
            a[i]      = (_Float16)a0[i];
            a[4 + i]  = (_Float16)a1[i];
            a[8 + i]  = (_Float16)a2[i];
            a[12 + i] = (_Float16)a3[i];
        }
#pragma unroll
        for (int ht = 0; ht < 8; ++ht)
            acc[ht] = __builtin_amdgcn_wmma_f32_16x16x32_f16(
                false, a, false, bf[ht], (short)0, acc[ht], false, false);
    }

    // D layout: VGPR r -> token row r0 + r + 8*hi, column h = ht*16 + m.
#pragma unroll
    for (int ht = 0; ht < 8; ++ht) {
        const int   col  = ht * 16 + m;
        const float bcol = bias[col];
        if (mode == 2) {
            const int b  = r0 / TSEQ;
            const int t0 = r0 % TSEQ;
            _Float16* vt = outbuf + (size_t)b * HEAD * TSEQ + (size_t)col * TSEQ;
#pragma unroll
            for (int r = 0; r < 8; ++r)
                vt[t0 + r + 8 * hi] = (_Float16)(acc[ht][r] + bcol);
        } else {
#pragma unroll
            for (int r = 0; r < 8; ++r) {
                float v = acc[ht][r] + bcol;
                if (mode == 0) v *= qscale;      // fold H^-1/2 into Q
                outbuf[(size_t)(r0 + r + 8 * hi) * HEAD + col] = (_Float16)v;
            }
        }
    }
}

// ---------------------------------------------------------------------------
// Flash attention: one wave owns 16 query rows, streams keys in chunks of 32.
//   S^T = K_tile * Q^T          (softmax reduction mostly intra-lane)
//   O^T += V^T_tile * P^T       (8 head-dim tiles of 16)
__global__ void attn_kernel(const _Float16* __restrict__ Qb,   // [TOKENS,128]
                            const _Float16* __restrict__ Kb,   // [TOKENS,128]
                            const _Float16* __restrict__ Vt,   // [B,128,TSEQ]
                            float* __restrict__ out) {         // [TOKENS,128]
    const int lane  = threadIdx.x & 31;
    const int wave  = threadIdx.x >> 5;
    const int qt    = blockIdx.x * 4 + wave;     // 0..1023 query tiles
    const int m     = lane & 15;
    const int hi    = lane >> 4;
    const int batch = qt / (TSEQ / 16);
    const int q0    = qt * 16;                   // global token row
    const size_t kb0 = (size_t)batch * TSEQ;

    // Q^T B-fragments for the 4 K-dim chunks (lane's column = query m)
    const _Float16* qrow = Qb + (size_t)(q0 + m) * HEAD;
    v16h bq[4];
#pragma unroll
    for (int cc = 0; cc < 4; ++cc)
        bq[cc] = load_fragB_f16(qrow + cc * 32, hi);

    v8f  o[8] = {};
    float mrun = -1e30f, lrun = 0.0f;
    const _Float16* vbase = Vt + (size_t)batch * HEAD * TSEQ + (size_t)m * TSEQ;
    const _Float16* kbase = Kb + (kb0 + m) * HEAD;

#pragma unroll 1
    for (int k0 = 0; k0 < TSEQ; k0 += 32) {
        const _Float16* krow0 = kbase + (size_t)k0 * HEAD;
        const _Float16* krow1 = krow0 + 16 * HEAD;

        // ---- issue all loads for this iteration up-front:
        // 8 K fragments (feed the S WMMAs) + 8 V fragments (only needed by the
        // O WMMAs, so their latency hides behind S-WMMA + softmax VALU work)
        v16h ak0[4], ak1[4];
#pragma unroll
        for (int cc = 0; cc < 4; ++cc) ak0[cc] = load_fragA_f16(krow0 + cc * 32, hi);
#pragma unroll
        for (int cc = 0; cc < 4; ++cc) ak1[cc] = load_fragA_f16(krow1 + cc * 32, hi);

        v16h av[8];
#pragma unroll
        for (int ht = 0; ht < 8; ++ht)
            av[ht] = load_fragA_f16(vbase + (size_t)ht * 16 * TSEQ + k0, hi);

        // prefetch next chunk's K rows (global_prefetch_b8; speculative)
        __builtin_prefetch(krow0 + 32 * HEAD, 0, 0);
        __builtin_prefetch(krow1 + 32 * HEAD, 0, 0);

        v8f s0 = {}, s1 = {};
#pragma unroll
        for (int cc = 0; cc < 4; ++cc)
            s0 = __builtin_amdgcn_wmma_f32_16x16x32_f16(
                false, ak0[cc], false, bq[cc], (short)0, s0, false, false);
#pragma unroll
        for (int cc = 0; cc < 4; ++cc)
            s1 = __builtin_amdgcn_wmma_f32_16x16x32_f16(
                false, ak1[cc], false, bq[cc], (short)0, s1, false, false);

        // Online softmax: this lane (query m) holds 16 of 32 key scores:
        //   lo-half: keys k0+0..7 (s0), k0+16..23 (s1)
        //   hi-half: keys k0+8..15 (s0), k0+24..31 (s1)
        float mloc = s0[0];
#pragma unroll
        for (int i = 1; i < 8; ++i) mloc = fmaxf(mloc, s0[i]);
#pragma unroll
        for (int i = 0; i < 8; ++i) mloc = fmaxf(mloc, s1[i]);
        float mnew = fmaxf(fmaxf(mloc, __shfl_xor(mloc, 16, 32)), mrun);
        float corr = __expf(mrun - mnew);
        mrun = mnew;

        float p0[8], p1[8], psum = 0.0f;
#pragma unroll
        for (int i = 0; i < 8; ++i) {
            p0[i] = __expf(s0[i] - mnew);
            p1[i] = __expf(s1[i] - mnew);
            psum += p0[i] + p1[i];
        }
        lrun = lrun * corr + psum;

#pragma unroll
        for (int t = 0; t < 8; ++t)
#pragma unroll
            for (int r = 0; r < 8; ++r) o[t][r] *= corr;

        // P^T B-fragment: lane needs keys (hi ? 16..31 : 0..15) contiguous.
        // Exchange the 8 values the partner half-wave owns.
        v16h pb;
#pragma unroll
        for (int i = 0; i < 8; ++i) {
            float send = hi ? p0[i] : p1[i];
            float recv = __shfl_xor(send, 16, 32);
            pb[i]     = (_Float16)(hi ? recv  : p0[i]);
            pb[8 + i] = (_Float16)(hi ? p1[i] : recv);
        }

        // O^T += V^T_chunk * P^T over 8 head-dim tiles
#pragma unroll
        for (int ht = 0; ht < 8; ++ht)
            o[ht] = __builtin_amdgcn_wmma_f32_16x16x32_f16(
                false, av[ht], false, pb, (short)0, o[ht], false, false);
    }

    float ltot = lrun + __shfl_xor(lrun, 16, 32);
    float inv  = 1.0f / ltot;
    // D layout of O^T: lane's column = query m, row h = ht*16 + r + 8*hi
    float* orow = out + (size_t)(q0 + m) * HEAD;
#pragma unroll
    for (int ht = 0; ht < 8; ++ht)
#pragma unroll
        for (int r = 0; r < 8; ++r)
            orow[ht * 16 + r + 8 * hi] = o[ht][r] * inv;
}

// ---------------------------------------------------------------------------
extern "C" void kernel_launch(void* const* d_in, const int* in_sizes, int n_in,
                              void* d_out, int out_size, void* d_ws, size_t ws_size,
                              hipStream_t stream) {
    (void)in_sizes; (void)n_in; (void)out_size; (void)ws_size;
    const float* x  = (const float*)d_in[0];
    const float* Wk = (const float*)d_in[1];
    const float* bk = (const float*)d_in[2];
    const float* Wq = (const float*)d_in[3];
    const float* bq = (const float*)d_in[4];
    const float* Wv = (const float*)d_in[5];
    const float* bv = (const float*)d_in[6];
    float* out = (float*)d_out;

    // Workspace layout (halfs): 3 f16 weight copies, then Q, K, V^T (~12.75 MB)
    _Float16* ws    = (_Float16*)d_ws;
    _Float16* wk16  = ws;
    _Float16* wq16  = ws + 131072;
    _Float16* wv16  = ws + 262144;
    _Float16* Qbuf  = ws + 393216;
    _Float16* Kbuf  = Qbuf + (size_t)TOKENS * HEAD;
    _Float16* Vtbuf = Kbuf + (size_t)TOKENS * HEAD;

    f32_to_f16_kernel<<<512, 256, 0, stream>>>(Wk, wk16, 131072);
    f32_to_f16_kernel<<<512, 256, 0, stream>>>(Wq, wq16, 131072);
    f32_to_f16_kernel<<<512, 256, 0, stream>>>(Wv, wv16, 131072);

    const float scale = 0.08838834764831845f;   // 128^-0.5 folded into Q
    qkv_proj_kernel<<<256, 128, 0, stream>>>(x, wq16, bq, Qbuf, 0, scale);
    qkv_proj_kernel<<<256, 128, 0, stream>>>(x, wk16, bk, Kbuf, 1, 0.0f);
    qkv_proj_kernel<<<256, 128, 0, stream>>>(x, wv16, bv, Vtbuf, 2, 0.0f);

    attn_kernel<<<256, 128, 0, stream>>>(Qbuf, Kbuf, Vtbuf, out);
}
